// InnerIteration_24507083391218
// MI455X (gfx1250) — matrix operand order repaired and checked by reading, not compiled, not run on hardware
//
#include <hip/hip_runtime.h>

#define DDIM 128
#define GDIM 32
#define KLIT 8
#define MT   2            // row-tiles (16 rows each) per block -> M = 32

typedef __attribute__((ext_vector_type(16))) __bf16          v16bf;
typedef __attribute__((ext_vector_type(16))) unsigned short  v16us;
typedef __attribute__((ext_vector_type(8)))  float           v8f;

static __device__ __forceinline__ unsigned short f32_to_bf16(float f) {
  unsigned u = __builtin_bit_cast(unsigned, f);
  unsigned r = u + 0x7FFFu + ((u >> 16) & 1u);   // round-to-nearest-even
  return (unsigned short)(r >> 16);
}

static __device__ __forceinline__ v8f wmma_bf16(v16us a, v16us b, v8f c) {
  return __builtin_amdgcn_wmma_f32_16x16x32_bf16(
      false, __builtin_bit_cast(v16bf, a),
      false, __builtin_bit_cast(v16bf, b),
      (short)0, c, false, false);
}

// A fragment from LDS: A is [16][lda] bf16 row-major, k-step base kb.
// Layout (contiguous, matches B): lanes 0-15 hold K=kb+0..15,
// lanes 16-31 hold K=kb+16..31, row M = lane&15.
static __device__ __forceinline__ v16us load_a_frag(const unsigned short* A, int lda,
                                                    int lane, int kb) {
  const int m  = lane & 15;
  const int hi = (lane >> 4) & 1;
  const unsigned short* row = A + m * lda + kb + hi * 16;
  v16us a;
#pragma unroll
  for (int e = 0; e < 8; ++e) a[e] = row[e];
#pragma unroll
  for (int e = 0; e < 8; ++e) a[8 + e] = row[8 + e];
  return a;
}

// B fragment from global pre-transposed weight WT[n][k] (n-major, row len ldk).
// Lane holds column N = n0+(lane&15); lanes 0-15 K=kb+0..15, lanes 16-31
// K=kb+16..31 -> one 32B contiguous run per lane.
static __device__ __forceinline__ v16us load_b_frag(const unsigned short* WT, int ldk,
                                                    int lane, int n0, int kb) {
  const int n  = n0 + (lane & 15);
  const int hi = (lane >> 4) & 1;
  const unsigned short* p = WT + (size_t)n * ldk + kb + hi * 16;
  v16us b;
#pragma unroll
  for (int e = 0; e < 16; ++e) b[e] = p[e];
  return b;
}

static __device__ __forceinline__ float sigmoidf(float x) {
  return 1.0f / (1.0f + __expf(-x));
}

static __device__ __forceinline__ void store8_bf16(unsigned short* d, float4 a, float4 b) {
  d[0] = f32_to_bf16(a.x); d[1] = f32_to_bf16(a.y); d[2] = f32_to_bf16(a.z); d[3] = f32_to_bf16(a.w);
  d[4] = f32_to_bf16(b.x); d[5] = f32_to_bf16(b.y); d[6] = f32_to_bf16(b.z); d[7] = f32_to_bf16(b.w);
}
static __device__ __forceinline__ void store8_f32(float* d, float4 a, float4 b) {
  d[0] = a.x; d[1] = a.y; d[2] = a.z; d[3] = a.w;
  d[4] = b.x; d[5] = b.y; d[6] = b.z; d[7] = b.w;
}

// ---------------------------------------------------------------------------
// Prep: W (K x 128 f32, row-major) -> WT (128 x K bf16, n-major / transposed)
// ---------------------------------------------------------------------------
__global__ __launch_bounds__(256) void prep_wT_bf16(const float* __restrict__ W,
                                                    unsigned short* __restrict__ WT,
                                                    int K) {
  int i = blockIdx.x * blockDim.x + threadIdx.x;
  if (i >= K * DDIM) return;
  int k = i / DDIM, n = i % DDIM;
  WT[(size_t)n * K + k] = f32_to_bf16(W[(size_t)k * DDIM + n]);
}

// ---------------------------------------------------------------------------
// Clause stage: 32 clauses/block, 256 threads (8 waves), 2 row-tiles/wave.
//   sum_pos/sum_neg gather -> (sum_pos + sum_neg@Wn + nneg*bn) @ Wv ->
//   tanh -> atomic scatter into agg; cnt counts occurrences.
// ---------------------------------------------------------------------------
__global__ __launch_bounds__(256) void clause_kernel(
    const float* __restrict__ variables,
    const int*   __restrict__ lits,
    const int*   __restrict__ neg,
    const unsigned short* __restrict__ WnT,
    const unsigned short* __restrict__ WvT,
    const float* __restrict__ bn,
    const float* __restrict__ bv,
    float* __restrict__ agg,
    float* __restrict__ cnt) {
  __shared__ int   s_lits[16 * MT][KLIT];
  __shared__ int   s_neg [16 * MT][KLIT];
  __shared__ float s_nneg[16 * MT];
  __shared__ float s_pos [16 * MT][DDIM];
  __shared__ unsigned short s_aneg [16 * MT][DDIM];
  __shared__ unsigned short s_acomb[16 * MT][DDIM];

  const int tid = threadIdx.x;
  const size_t c0 = (size_t)blockIdx.x * (16 * MT);

  // ---- literal/neg load: 256 threads cover 32 rows x 8 lits exactly ----
  {
    const int r  = tid >> 3;      // 0..31
    const int kq = tid & 7;       // 0..7
    size_t o = (c0 + r) * KLIT + kq;
    int idx = lits[o];
    int ng  = neg[o];
    s_lits[r][kq] = idx;
    s_neg [r][kq] = ng;
    unsafeAtomicAdd(&cnt[idx], 1.0f);          // global_atomic_add_f32
  }
  __syncthreads();

  if (tid < 16 * MT) {
    int nn = 0;
#pragma unroll
    for (int k = 0; k < KLIT; ++k) nn += s_neg[tid][k];
    s_nneg[tid] = (float)nn;
  }

  // ---- gather: thread owns 8 dims of 2 clause rows ----
  const int ci    = tid >> 4;     // 0..15
  const int p     = tid & 15;     // 0..15
  const int dbase = p * 8;
#pragma unroll
  for (int half = 0; half < MT; ++half) {
    const int rr = half * 16 + ci;
    float4 pa = {0, 0, 0, 0}, pb = {0, 0, 0, 0};
    float4 na = {0, 0, 0, 0}, nb = {0, 0, 0, 0};
#pragma unroll
    for (int k = 0; k < KLIT; ++k) {
      const float* src = variables + (size_t)s_lits[rr][k] * DDIM + dbase;
      if (k + 1 < KLIT)
        __builtin_prefetch(variables + (size_t)s_lits[rr][k + 1] * DDIM + dbase, 0, 0);
      float4 x0 = ((const float4*)src)[0];
      float4 x1 = ((const float4*)src)[1];
      if (s_neg[rr][k]) {
        na.x += x0.x; na.y += x0.y; na.z += x0.z; na.w += x0.w;
        nb.x += x1.x; nb.y += x1.y; nb.z += x1.z; nb.w += x1.w;
      } else {
        pa.x += x0.x; pa.y += x0.y; pa.z += x0.z; pa.w += x0.w;
        pb.x += x1.x; pb.y += x1.y; pb.z += x1.z; pb.w += x1.w;
      }
    }
    store8_f32 (&s_pos [rr][dbase], pa, pb);
    store8_bf16(&s_aneg[rr][dbase], na, nb);
  }
  __syncthreads();

  // ---- WMMA: wave owns a 16-col tile; B fragment reused across MT row tiles
  const int wave = tid >> 5;
  const int lane = tid & 31;
  const int n0   = wave * 16;
  const int n    = n0 + (lane & 15);
  const int hi   = (lane >> 4) & 1;

  // GEMM1: T = sum_neg @ Wn ; combined = sum_pos + T + nneg*bn
  v8f acc[MT];
#pragma unroll
  for (int mt = 0; mt < MT; ++mt)
#pragma unroll
    for (int j = 0; j < 8; ++j) acc[mt][j] = 0.f;
#pragma unroll
  for (int kk = 0; kk < 4; ++kk) {
    v16us b = load_b_frag(WnT, DDIM, lane, n0, kk * 32);
#pragma unroll
    for (int mt = 0; mt < MT; ++mt) {
      v16us a = load_a_frag(&s_aneg[mt * 16][0], DDIM, lane, kk * 32);
      acc[mt] = wmma_bf16(a, b, acc[mt]);
    }
  }
  const float bnv = bn[n];
#pragma unroll
  for (int mt = 0; mt < MT; ++mt)
#pragma unroll
    for (int j = 0; j < 8; ++j) {
      int m = mt * 16 + j + hi * 8;
      float v = acc[mt][j] + s_pos[m][n] + s_nneg[m] * bnv;
      s_acomb[m][n] = f32_to_bf16(v);
    }
  __syncthreads();

  // GEMM2: clause_emb = tanh(combined @ Wv + bv), then scatter to agg.
#pragma unroll
  for (int mt = 0; mt < MT; ++mt)
#pragma unroll
    for (int j = 0; j < 8; ++j) acc[mt][j] = 0.f;
#pragma unroll
  for (int kk = 0; kk < 4; ++kk) {
    v16us b = load_b_frag(WvT, DDIM, lane, n0, kk * 32);
#pragma unroll
    for (int mt = 0; mt < MT; ++mt) {
      v16us a = load_a_frag(&s_acomb[mt * 16][0], DDIM, lane, kk * 32);
      acc[mt] = wmma_bf16(a, b, acc[mt]);
    }
  }
  const float bvv = bv[n];
#pragma unroll
  for (int mt = 0; mt < MT; ++mt)
#pragma unroll
    for (int j = 0; j < 8; ++j) {
      int m = mt * 16 + j + hi * 8;
      float e = tanhf(acc[mt][j] + bvv);
#pragma unroll
      for (int k = 0; k < KLIT; ++k) {
        unsafeAtomicAdd(&agg[(size_t)s_lits[m][k] * DDIM + n], e);
      }
    }
}

// ---------------------------------------------------------------------------
// Variable stage: 32 variables/block, 256 threads (8 waves), 2 row-tiles/wave.
//   agg2 = tanh(agg@Wc+bc); new = tanh([ground,agg2]@Wg+bg);
//   av = cnt>0 ? new : var; GRU gates z/r/h; out = (1-z)*var + z*h.
// ---------------------------------------------------------------------------
__global__ __launch_bounds__(256) void var_kernel(
    const float* __restrict__ variables,
    const float* __restrict__ ground,
    const float* __restrict__ agg,
    const float* __restrict__ cnt,
    const unsigned short* __restrict__ WcT,
    const unsigned short* __restrict__ WgT,
    const unsigned short* __restrict__ WzT,
    const unsigned short* __restrict__ UzT,
    const unsigned short* __restrict__ WrT,
    const unsigned short* __restrict__ UrT,
    const unsigned short* __restrict__ WwT,
    const unsigned short* __restrict__ UuT,
    const float* __restrict__ bc, const float* __restrict__ bg,
    const float* __restrict__ bz, const float* __restrict__ br,
    const float* __restrict__ bu,
    float* __restrict__ out) {
  __shared__ unsigned short s_agg [16 * MT][DDIM];
  __shared__ unsigned short s_a2  [16 * MT][GDIM + DDIM];  // [0:32)=ground, [32:160)=agg2
  __shared__ unsigned short s_var [16 * MT][DDIM];
  __shared__ unsigned short s_av  [16 * MT][DDIM];
  __shared__ unsigned short s_rv  [16 * MT][DDIM];
  __shared__ float          s_varf[16 * MT][DDIM];
  __shared__ float          s_pred[16 * MT];

  const int tid = threadIdx.x;
  const int ci  = tid >> 4;
  const int p   = tid & 15;
  const size_t v0 = (size_t)blockIdx.x * (16 * MT);
  const int dbase = p * 8;

#pragma unroll
  for (int half = 0; half < MT; ++half) {
    const int rr = half * 16 + ci;
    const float* vp = variables + (v0 + rr) * DDIM + dbase;
    float4 x0 = ((const float4*)vp)[0], x1 = ((const float4*)vp)[1];
    store8_f32 (&s_varf[rr][dbase], x0, x1);
    store8_bf16(&s_var [rr][dbase], x0, x1);
    const float* ap = agg + (v0 + rr) * DDIM + dbase;
    float4 a0 = ((const float4*)ap)[0], a1 = ((const float4*)ap)[1];
    store8_bf16(&s_agg[rr][dbase], a0, a1);
    if (p < 4) {
      const float* gp = ground + (v0 + rr) * GDIM + dbase;
      float4 g0 = ((const float4*)gp)[0], g1 = ((const float4*)gp)[1];
      store8_bf16(&s_a2[rr][dbase], g0, g1);
    }
    if (p == 0) s_pred[rr] = cnt[v0 + rr];
  }
  __syncthreads();

  const int wave = tid >> 5;
  const int lane = tid & 31;
  const int n0   = wave * 16;
  const int n    = n0 + (lane & 15);
  const int hi   = (lane >> 4) & 1;
  const int LDA2 = GDIM + DDIM;

  // agg2 = tanh(agg @ Wc + bc) -> columns 32..159 of s_a2
  v8f acc[MT];
#pragma unroll
  for (int mt = 0; mt < MT; ++mt)
#pragma unroll
    for (int j = 0; j < 8; ++j) acc[mt][j] = 0.f;
#pragma unroll
  for (int kk = 0; kk < 4; ++kk) {
    v16us b = load_b_frag(WcT, DDIM, lane, n0, kk * 32);
#pragma unroll
    for (int mt = 0; mt < MT; ++mt) {
      v16us a = load_a_frag(&s_agg[mt * 16][0], DDIM, lane, kk * 32);
      acc[mt] = wmma_bf16(a, b, acc[mt]);
    }
  }
  {
    float bcv = bc[n];
#pragma unroll
    for (int mt = 0; mt < MT; ++mt)
#pragma unroll
      for (int j = 0; j < 8; ++j) {
        int m = mt * 16 + j + hi * 8;
        s_a2[m][GDIM + n] = f32_to_bf16(tanhf(acc[mt][j] + bcv));
      }
  }
  __syncthreads();

  // new_emb = tanh([ground, agg2] @ Wg + bg); av = pred ? new_emb : var
#pragma unroll
  for (int mt = 0; mt < MT; ++mt)
#pragma unroll
    for (int j = 0; j < 8; ++j) acc[mt][j] = 0.f;
#pragma unroll
  for (int kk = 0; kk < 5; ++kk) {          // K = 160
    v16us b = load_b_frag(WgT, LDA2, lane, n0, kk * 32);
#pragma unroll
    for (int mt = 0; mt < MT; ++mt) {
      v16us a = load_a_frag(&s_a2[mt * 16][0], LDA2, lane, kk * 32);
      acc[mt] = wmma_bf16(a, b, acc[mt]);
    }
  }
  {
    float bgv = bg[n];
#pragma unroll
    for (int mt = 0; mt < MT; ++mt)
#pragma unroll
      for (int j = 0; j < 8; ++j) {
        int m = mt * 16 + j + hi * 8;
        float e  = tanhf(acc[mt][j] + bgv);
        float av = (s_pred[m] > 0.f) ? e : s_varf[m][n];
        s_av[m][n] = f32_to_bf16(av);
      }
  }
  __syncthreads();

  // z gate (kept in registers per row-tile)
  v8f z[MT];
#pragma unroll
  for (int mt = 0; mt < MT; ++mt)
#pragma unroll
    for (int j = 0; j < 8; ++j) acc[mt][j] = 0.f;
#pragma unroll
  for (int kk = 0; kk < 4; ++kk) {
    v16us b = load_b_frag(WzT, DDIM, lane, n0, kk * 32);
#pragma unroll
    for (int mt = 0; mt < MT; ++mt) {
      v16us a = load_a_frag(&s_av[mt * 16][0], DDIM, lane, kk * 32);
      acc[mt] = wmma_bf16(a, b, acc[mt]);
    }
  }
#pragma unroll
  for (int kk = 0; kk < 4; ++kk) {
    v16us b = load_b_frag(UzT, DDIM, lane, n0, kk * 32);
#pragma unroll
    for (int mt = 0; mt < MT; ++mt) {
      v16us a = load_a_frag(&s_var[mt * 16][0], DDIM, lane, kk * 32);
      acc[mt] = wmma_bf16(a, b, acc[mt]);
    }
  }
  {
    float bzv = bz[n];
#pragma unroll
    for (int mt = 0; mt < MT; ++mt)
#pragma unroll
      for (int j = 0; j < 8; ++j) z[mt][j] = sigmoidf(acc[mt][j] + bzv);
  }

  // r gate -> rv = r * var
#pragma unroll
  for (int mt = 0; mt < MT; ++mt)
#pragma unroll
    for (int j = 0; j < 8; ++j) acc[mt][j] = 0.f;
#pragma unroll
  for (int kk = 0; kk < 4; ++kk) {
    v16us b = load_b_frag(WrT, DDIM, lane, n0, kk * 32);
#pragma unroll
    for (int mt = 0; mt < MT; ++mt) {
      v16us a = load_a_frag(&s_av[mt * 16][0], DDIM, lane, kk * 32);
      acc[mt] = wmma_bf16(a, b, acc[mt]);
    }
  }
#pragma unroll
  for (int kk = 0; kk < 4; ++kk) {
    v16us b = load_b_frag(UrT, DDIM, lane, n0, kk * 32);
#pragma unroll
    for (int mt = 0; mt < MT; ++mt) {
      v16us a = load_a_frag(&s_var[mt * 16][0], DDIM, lane, kk * 32);
      acc[mt] = wmma_bf16(a, b, acc[mt]);
    }
  }
  {
    float brv = br[n];
#pragma unroll
    for (int mt = 0; mt < MT; ++mt)
#pragma unroll
      for (int j = 0; j < 8; ++j) {
        int m = mt * 16 + j + hi * 8;
        float r = sigmoidf(acc[mt][j] + brv);
        s_rv[m][n] = f32_to_bf16(r * s_varf[m][n]);
      }
  }
  __syncthreads();

  // h_tilda and output
#pragma unroll
  for (int mt = 0; mt < MT; ++mt)
#pragma unroll
    for (int j = 0; j < 8; ++j) acc[mt][j] = 0.f;
#pragma unroll
  for (int kk = 0; kk < 4; ++kk) {
    v16us b = load_b_frag(WwT, DDIM, lane, n0, kk * 32);
#pragma unroll
    for (int mt = 0; mt < MT; ++mt) {
      v16us a = load_a_frag(&s_av[mt * 16][0], DDIM, lane, kk * 32);
      acc[mt] = wmma_bf16(a, b, acc[mt]);
    }
  }
#pragma unroll
  for (int kk = 0; kk < 4; ++kk) {
    v16us b = load_b_frag(UuT, DDIM, lane, n0, kk * 32);
#pragma unroll
    for (int mt = 0; mt < MT; ++mt) {
      v16us a = load_a_frag(&s_rv[mt * 16][0], DDIM, lane, kk * 32);
      acc[mt] = wmma_bf16(a, b, acc[mt]);
    }
  }
  {
    float buv = bu[n];
#pragma unroll
    for (int mt = 0; mt < MT; ++mt)
#pragma unroll
      for (int j = 0; j < 8; ++j) {
        int m = mt * 16 + j + hi * 8;
        float h  = tanhf(acc[mt][j] + buv);
        float vf = s_varf[m][n];
        out[(v0 + m) * DDIM + n] = (1.f - z[mt][j]) * vf + z[mt][j] * h;
      }
  }
}

// ---------------------------------------------------------------------------
extern "C" void kernel_launch(void* const* d_in, const int* in_sizes, int n_in,
                              void* d_out, int out_size, void* d_ws, size_t ws_size,
                              hipStream_t stream) {
  const float* variables = (const float*)d_in[0];
  const float* ground    = (const float*)d_in[1];
  const int*   lits      = (const int*)d_in[2];
  const int*   neg       = (const int*)d_in[3];
  const float* Wn = (const float*)d_in[4];  const float* bn = (const float*)d_in[5];
  const float* Wv = (const float*)d_in[6];  const float* bv = (const float*)d_in[7];
  const float* Wc = (const float*)d_in[8];  const float* bc = (const float*)d_in[9];
  const float* Wg = (const float*)d_in[10]; const float* bg = (const float*)d_in[11];
  const float* Wz = (const float*)d_in[12]; const float* Uz = (const float*)d_in[13];
  const float* bz = (const float*)d_in[14];
  const float* Wr = (const float*)d_in[15]; const float* Ur = (const float*)d_in[16];
  const float* br = (const float*)d_in[17];
  const float* Ww = (const float*)d_in[18]; const float* Uu = (const float*)d_in[19];
  const float* bu = (const float*)d_in[20];

  const int N = in_sizes[0] / DDIM;
  const int C = in_sizes[2] / KLIT;

  // Workspace layout: [agg f32 N*128][cnt f32 N][bf16 transposed weights]
  char* ws = (char*)d_ws;
  float* agg = (float*)ws;
  float* cnt = (float*)(ws + (size_t)N * DDIM * sizeof(float));
  unsigned short* wt = (unsigned short*)(ws + (size_t)N * DDIM * sizeof(float)
                                            + (size_t)N * sizeof(float));
  const size_t SQ = (size_t)DDIM * DDIM;          // 16384
  unsigned short* WnT = wt;             wt += SQ;
  unsigned short* WvT = wt;             wt += SQ;
  unsigned short* WcT = wt;             wt += SQ;
  unsigned short* WgT = wt;             wt += (size_t)(GDIM + DDIM) * DDIM;  // 160*128
  unsigned short* WzT = wt;             wt += SQ;
  unsigned short* UzT = wt;             wt += SQ;
  unsigned short* WrT = wt;             wt += SQ;
  unsigned short* UrT = wt;             wt += SQ;
  unsigned short* WwT = wt;             wt += SQ;
  unsigned short* UuT = wt;

  // Zero accumulators (agg + cnt are contiguous).
  hipMemsetAsync(agg, 0, (size_t)N * DDIM * sizeof(float) + (size_t)N * sizeof(float),
                 stream);

  // Transpose-convert weights to bf16.
  const int g128 = (DDIM * DDIM + 255) / 256;
  const int g160 = ((GDIM + DDIM) * DDIM + 255) / 256;
  prep_wT_bf16<<<g128, 256, 0, stream>>>(Wn, WnT, DDIM);
  prep_wT_bf16<<<g128, 256, 0, stream>>>(Wv, WvT, DDIM);
  prep_wT_bf16<<<g128, 256, 0, stream>>>(Wc, WcT, DDIM);
  prep_wT_bf16<<<g160, 256, 0, stream>>>(Wg, WgT, GDIM + DDIM);
  prep_wT_bf16<<<g128, 256, 0, stream>>>(Wz, WzT, DDIM);
  prep_wT_bf16<<<g128, 256, 0, stream>>>(Uz, UzT, DDIM);
  prep_wT_bf16<<<g128, 256, 0, stream>>>(Wr, WrT, DDIM);
  prep_wT_bf16<<<g128, 256, 0, stream>>>(Ur, UrT, DDIM);
  prep_wT_bf16<<<g128, 256, 0, stream>>>(Ww, WwT, DDIM);
  prep_wT_bf16<<<g128, 256, 0, stream>>>(Uu, UuT, DDIM);

  // Clause stage: gather + 2 WMMA GEMMs + atomic scatter.
  clause_kernel<<<C / (16 * MT), 256, 0, stream>>>(variables, lits, neg, WnT, WvT,
                                                   bn, bv, agg, cnt);

  // Variable stage: 8 WMMA GEMMs + GRU.
  var_kernel<<<N / (16 * MT), 256, 0, stream>>>(variables, ground, agg, cnt,
                                                WcT, WgT, WzT, UzT, WrT, UrT, WwT, UuT,
                                                bc, bg, bz, br, bu, (float*)d_out);
}
